// MultiDilatelocalAttention_41661182771557
// MI455X (gfx1250) — matrix operand order repaired and checked by reading, not compiled, not run on hardware
//
#include <hip/hip_runtime.h>

// ---------------------------------------------------------------------------
// MultiDilate local attention for MI455X (gfx1250, wave32, WMMA)
//   B=8, C=256, H=W=64, 8 heads x head_dim 32, 3x3 neighborhood, dil 1..4
// GEMMs: v_wmma_f32_16x16x32_bf16, weight tile staged in LDS (ds_load_b128),
// 32Mx64N per wave (8 WMMA per K-step vs 12 b128 loads).  Attention is fp32
// VALU (K=9 is not matrix-shaped).  qkv (100MB) stays resident in 192MB L2.
// ---------------------------------------------------------------------------

typedef __attribute__((ext_vector_type(16))) __bf16 v16bf;
typedef __attribute__((ext_vector_type(8)))  float  v8f;

#define BB   8
#define CC   256
#define HW   4096      // 64*64
#define WDIM 64
#define NH   8
#define HD   32
#define ATT_SCALE 0.17677669529663687f   // 32^-0.5

__device__ __forceinline__ unsigned short f2bf(float f) {
    union { float f; unsigned u; } x; x.f = f;
    unsigned r = x.u + 0x7FFFu + ((x.u >> 16) & 1u);   // round-to-nearest-even
    return (unsigned short)(r >> 16);
}

// ---------------------------------------------------------------------------
// x [b][c][hw] f32  ->  xt [b][hw][c] bf16   (32x32 LDS tile transpose)
// ---------------------------------------------------------------------------
__global__ __launch_bounds__(256) void x_to_bf16_t(const float* __restrict__ x,
                                                   unsigned short* __restrict__ xt) {
    __shared__ float sm[32][33];
    const int tx = threadIdx.x & 31;
    const int ty = threadIdx.x >> 5;          // 0..7
    const int b  = blockIdx.z;
    const int c0 = blockIdx.y * 32;
    const int h0 = blockIdx.x * 32;
#pragma unroll
    for (int r = 0; r < 4; ++r) {
        int c = ty + r * 8;
        sm[c][tx] = x[((size_t)b * CC + c0 + c) * HW + h0 + tx];
    }
    __syncthreads();
#pragma unroll
    for (int r = 0; r < 4; ++r) {
        int h = ty + r * 8;
        xt[((size_t)b * HW + h0 + h) * CC + c0 + tx] = f2bf(sm[tx][h]);
    }
}

// ---------------------------------------------------------------------------
// elementwise f32 -> bf16 (weights; [o][c] layout kept — K-contiguous rows)
// ---------------------------------------------------------------------------
__global__ void w_to_bf16(const float* __restrict__ w,
                          unsigned short* __restrict__ o, int n) {
    int i = blockIdx.x * blockDim.x + threadIdx.x;
    if (i < n) o[i] = f2bf(w[i]);
}

// ---------------------------------------------------------------------------
// bf16 GEMM via v_wmma_f32_16x16x32_bf16.
//   A  : [B][4096][256] bf16 (pixel-major, K contiguous)
//   Wt : [N][256]       bf16 (row o, K contiguous)
//   Out: [B][N][4096]   f32 (channel-major) = A*Wt^T + bias
// Block = 256 threads = 8 waves.  Block tile: 256(M) x 64(N).
// The 64x256 bf16 weight tile (32 KB) is staged in LDS once per block; each
// wave computes 32(M) x 64(N): per K-step(32) -> 2 A-frags (4 global b128),
// 4 B-frags (8 ds b128), 8 WMMA.
// Fragment layouts per CDNA5 ISA 7.12.2:
//   A lane L: M=L%16; elems 0..7 -> K = 8*(L/16)+0..7 ; elems 8..15 -> +16
//   B lane L: N=L%16; elems 0..7 -> K = 8*(L/16)+0..7 ; elems 8..15 -> +16 (same as A)
//   D vgpr r, lane L: (M = r + 8*(L/16), N = L%16)
// ---------------------------------------------------------------------------
__global__ __launch_bounds__(256) void gemm_bf16(const unsigned short* __restrict__ A,
                                                 const unsigned short* __restrict__ Wt,
                                                 const float* __restrict__ bias,
                                                 float* __restrict__ Out,
                                                 int Ntot) {
    __shared__ unsigned short smW[64 * CC];            // 32 KB weight tile [n_rel][k]

    const int lane  = threadIdx.x & 31;
    const int wid   = threadIdx.x >> 5;                // 0..7
    const int b     = blockIdx.z;
    const int m0    = blockIdx.x * 256 + wid * 32;
    const int n0    = blockIdx.y * 64;
    const int mrow  = lane & 15;
    const int khalf = lane >> 4;

    // ---- cooperative stage of the weight tile: 64 rows x 256 K, contiguous
    {
        const uint4* gw = (const uint4*)(Wt + (size_t)n0 * CC);
        uint4*       sw = (uint4*)smW;
#pragma unroll
        for (int i = 0; i < 8; ++i)
            sw[threadIdx.x + i * 256] = gw[threadIdx.x + i * 256];
    }
    __syncthreads();

    const unsigned short* Arow0 = A + ((size_t)b * HW + m0 + mrow) * CC;
    const unsigned short* Arow1 = Arow0 + 16 * CC;

    v8f acc[2][4] = {};
    union V16 { v16bf v; uint4 q[2]; };

    for (int k = 0; k < CC; k += 32) {
        V16 a0, a1;
        a0.q[0] = *(const uint4*)(Arow0 + k + khalf * 8);
        a0.q[1] = *(const uint4*)(Arow0 + k + 16 + khalf * 8);
        a1.q[0] = *(const uint4*)(Arow1 + k + khalf * 8);
        a1.q[1] = *(const uint4*)(Arow1 + k + 16 + khalf * 8);
#pragma unroll
        for (int f = 0; f < 4; ++f) {
            V16 bq;
            bq.q[0] = *(const uint4*)(smW + (f * 16 + mrow) * CC + k + khalf * 8);
            bq.q[1] = *(const uint4*)(smW + (f * 16 + mrow) * CC + k + 16 + khalf * 8);
            v16bf bw = bq.v;
            acc[0][f] = __builtin_amdgcn_wmma_f32_16x16x32_bf16(
                false, a0.v, false, bw, (short)0, acc[0][f], false, false);
            acc[1][f] = __builtin_amdgcn_wmma_f32_16x16x32_bf16(
                false, a1.v, false, bw, (short)0, acc[1][f], false, false);
        }
    }

#pragma unroll
    for (int f = 0; f < 4; ++f) {
        const int   n  = n0 + f * 16 + mrow;
        const float bv = bias[n];
#pragma unroll
        for (int g = 0; g < 2; ++g) {
            float* op = Out + ((size_t)b * Ntot + n) * HW + m0 + g * 16 + khalf * 8;
#pragma unroll
            for (int r = 0; r < 8; ++r) op[r] = acc[g][f][r] + bv;   // 32B contiguous
        }
    }
}

// ---------------------------------------------------------------------------
// Dilated 3x3 local attention, one thread per (b, head, pixel).
// Zero-pad semantics match the reference unfold: out-of-range neighbors keep
// logit = 0 (k=0 => q.k=0) and contribute v=0, but DO enter the softmax.
// Reads qkv [b][768][4096] f32 (coalesced over lanes; L2-resident).
// Writes ao [b][hw][256] bf16 (pixel-major -> GEMM2 A operand).
// ---------------------------------------------------------------------------
__global__ __launch_bounds__(256) void attn_kernel(const float* __restrict__ qkv,
                                                   unsigned short* __restrict__ ao) {
    const int b    = blockIdx.z;
    const int head = blockIdx.y;
    const int p    = blockIdx.x * 256 + threadIdx.x;
    const int py   = p >> 6;
    const int px   = p & 63;
    const int dil  = (head & 3) + 1;

    const float* base = qkv + (size_t)b * 3 * CC * HW;
    const float* qb   = base + (size_t)(head * HD) * HW + p;
    const float* kb   = base + (size_t)(CC + head * HD) * HW;
    const float* vb   = base + (size_t)(2 * CC + head * HD) * HW;

    float q[HD];
#pragma unroll
    for (int d = 0; d < HD; ++d) q[d] = qb[(size_t)d * HW];

    float logit[9];
    int   npix[9];
#pragma unroll
    for (int t = 0; t < 9; ++t) {
        const int iy = py + (t / 3 - 1) * dil;
        const int ix = px + (t % 3 - 1) * dil;
        if (iy >= 0 && iy < WDIM && ix >= 0 && ix < WDIM) {
            const int pn = iy * WDIM + ix;
            npix[t] = pn;
            float s = 0.f;
#pragma unroll
            for (int d = 0; d < HD; ++d) s += q[d] * kb[(size_t)d * HW + pn];
            logit[t] = s * ATT_SCALE;
        } else {
            npix[t]  = -1;
            logit[t] = 0.f;       // zero-padded k => logit 0, still in softmax
        }
    }

    float mx = logit[0];
#pragma unroll
    for (int t = 1; t < 9; ++t) mx = fmaxf(mx, logit[t]);
    float wsum = 0.f;
#pragma unroll
    for (int t = 0; t < 9; ++t) { logit[t] = __expf(logit[t] - mx); wsum += logit[t]; }
    const float inv = 1.f / wsum;

    float out[HD];
#pragma unroll
    for (int d = 0; d < HD; ++d) out[d] = 0.f;
#pragma unroll
    for (int t = 0; t < 9; ++t) {
        if (npix[t] >= 0) {
            const float wgt = logit[t] * inv;
            const int   pn  = npix[t];
#pragma unroll
            for (int d = 0; d < HD; ++d) out[d] += wgt * vb[(size_t)d * HW + pn];
        }
    }

    unsigned short* op = ao + ((size_t)b * HW + p) * CC + head * HD;
#pragma unroll
    for (int d = 0; d < HD; ++d) op[d] = f2bf(out[d]);   // 64B contiguous
}

// ---------------------------------------------------------------------------
extern "C" void kernel_launch(void* const* d_in, const int* in_sizes, int n_in,
                              void* d_out, int out_size, void* d_ws, size_t ws_size,
                              hipStream_t stream) {
    const float* x      = (const float*)d_in[0];
    const float* w_qkv  = (const float*)d_in[1];
    const float* b_qkv  = (const float*)d_in[2];
    const float* w_proj = (const float*)d_in[3];
    const float* b_proj = (const float*)d_in[4];

    char* ws = (char*)d_ws;
    float*          qkv = (float*)ws;          ws += (size_t)BB * 3 * CC * HW * 4; // 100.7 MB
    unsigned short* xt  = (unsigned short*)ws; ws += (size_t)BB * HW * CC * 2;     //  16.8 MB
    unsigned short* ao  = (unsigned short*)ws; ws += (size_t)BB * HW * CC * 2;     //  16.8 MB
    unsigned short* wq  = (unsigned short*)ws; ws += (size_t)3 * CC * CC * 2;      //   0.4 MB
    unsigned short* wp  = (unsigned short*)ws;                                     //   0.1 MB

    // 1) x -> bf16 pixel-major
    x_to_bf16_t<<<dim3(HW / 32, CC / 32, BB), 256, 0, stream>>>(x, xt);
    // 2) weights -> bf16
    w_to_bf16<<<dim3((3 * CC * CC + 255) / 256), 256, 0, stream>>>(w_qkv, wq, 3 * CC * CC);
    w_to_bf16<<<dim3((CC * CC + 255) / 256), 256, 0, stream>>>(w_proj, wp, CC * CC);
    // 3) QKV projection (WMMA): [B,4096,256] x [768,256]^T -> [B,768,4096]
    gemm_bf16<<<dim3(HW / 256, (3 * CC) / 64, BB), 256, 0, stream>>>(xt, wq, b_qkv, qkv, 3 * CC);
    // 4) multi-dilated local attention -> [B,4096,256] bf16
    attn_kernel<<<dim3(HW / 256, NH, BB), 256, 0, stream>>>(qkv, ao);
    // 5) output projection (WMMA): [B,4096,256] x [256,256]^T -> [B,256,4096]
    gemm_bf16<<<dim3(HW / 256, CC / 64, BB), 256, 0, stream>>>(ao, wp, b_proj, (float*)d_out, CC);
}
